// LSTM_64750926955020
// MI455X (gfx1250) — compile-verified, matrix operand order checked by reading, not run on hardware
//
#include <hip/hip_runtime.h>
#include <hip/hip_bf16.h>

typedef _Float16 h16;
typedef __attribute__((ext_vector_type(16))) _Float16 v16h;
typedef __attribute__((ext_vector_type(8)))  _Float16 v8h;
typedef __attribute__((ext_vector_type(8)))  float    v8f;
typedef __attribute__((ext_vector_type(4)))  unsigned int u32x4;
typedef __attribute__((ext_vector_type(8)))  int i32x8;
typedef __attribute__((ext_vector_type(4)))  int i32x4;

#define T_  512
#define B_  256
#define I_  120
#define H_  128
#define G_  512   // 4*H
#define KP  128   // K padded to 128
#define TB  (T_*B_)
#define D1_ 512
#define OUT_ 7

union V16 { v16h v; v8h h[2]; };

__device__ __forceinline__ float sigmoidf_(float x) { return 1.0f / (1.0f + __expf(-x)); }

// ---------------------------------------------------------------------------
// TDM: DMA one contiguous 16KB tile (2048 x 8B) from global into LDS.
// D# group0: count=1 | lds_addr | global_addr | type=2
// D# group1: data_size=3(8B), tensor_dim0=2048, tensor_dim1=1,
//            tile_dim0=2048, tile_dim1=1, stride0=2048.
// ---------------------------------------------------------------------------
__device__ __forceinline__ void tdm_load_16kb(unsigned long long ga, unsigned lds_byte_off)
{
    u32x4 g0;
    g0[0] = 1u;                                                   // count=1 (valid user D#)
    g0[1] = lds_byte_off;                                         // lds_addr
    g0[2] = (unsigned)(ga & 0xffffffffull);                       // global_addr[31:0]
    g0[3] = (unsigned)((ga >> 32) & 0x01ffffffull) | (2u << 30);  // addr[56:32] | type=2
    i32x8 g1;
    g1[0] = 0x00030000;            // workgroup_mask=0, data_size=3 (8 bytes)
    g1[1] = (int)(2048u << 16);    // tensor_dim0[15:0] at bits[63:48]
    g1[2] = (int)(1u << 16);       // tensor_dim0[31:16]=0, tensor_dim1[15:0]=1
    g1[3] = (int)(2048u << 16);    // tensor_dim1[31:16]=0, tile_dim0=2048
    g1[4] = 1;                     // tile_dim1=1, tile_dim2=0
    g1[5] = 2048;                  // tensor_dim0_stride low 32
    g1[6] = 0;                     // stride hi / tensor_dim1_stride lo
    g1[7] = 0;
    i32x4 z4 = {0, 0, 0, 0};
#if defined(__clang_major__) && (__clang_major__ >= 23)
    i32x8 z8 = {0, 0, 0, 0, 0, 0, 0, 0};
    __builtin_amdgcn_tensor_load_to_lds(g0, g1, z4, z4, z8, 0);
#else
    __builtin_amdgcn_tensor_load_to_lds(g0, g1, z4, z4, 0);
#endif
}

// ---------------------------------------------------------------------------
// Pack weights / inputs to f16 (K padded to 128), fuse biases.
// ---------------------------------------------------------------------------
__global__ void pack_kernel(const float* __restrict__ x,      // [T,B,I]
                            const float* __restrict__ Wih0,   // [G,I]
                            const float* __restrict__ Whh0,   // [G,H]
                            const float* __restrict__ Wih1,   // [G,H]
                            const float* __restrict__ Whh1,   // [G,H]
                            const float* __restrict__ bih0, const float* __restrict__ bhh0,
                            const float* __restrict__ bih1, const float* __restrict__ bhh1,
                            h16* __restrict__ x16,            // [TB,KP]
                            h16* __restrict__ wih0p, h16* __restrict__ whh0p,
                            h16* __restrict__ wih1p, h16* __restrict__ whh1p,
                            float* __restrict__ bias0, float* __restrict__ bias1)
{
    const long stride = (long)gridDim.x * blockDim.x;
    const long i0 = (long)blockIdx.x * blockDim.x + threadIdx.x;
    for (long e = i0; e < (long)TB * KP; e += stride) {
        const int f = (int)(e & (KP - 1));
        const long r = e >> 7;
        x16[e] = (f < I_) ? (h16)x[r * I_ + f] : (h16)0.0f;
    }
    for (long e = i0; e < (long)G_ * KP; e += stride) {
        const int g = (int)(e >> 7), f = (int)(e & (KP - 1));
        wih0p[e] = (f < I_) ? (h16)Wih0[g * I_ + f] : (h16)0.0f;
        whh0p[e] = (h16)Whh0[e];
        wih1p[e] = (h16)Wih1[e];
        whh1p[e] = (h16)Whh1[e];
    }
    for (long e = i0; e < G_; e += stride) {
        bias0[e] = bih0[e] + bhh0[e];
        bias1[e] = bih1[e] + bhh1[e];
    }
}

// ---------------------------------------------------------------------------
// Input projection GEMM: Y[M,G] (f16) = X[M,KP] @ W[G,KP]^T + bias[G]
// One wave computes a 16x64 output tile (4 WMMA D-tiles sharing A).
// ---------------------------------------------------------------------------
__global__ void proj_kernel(const h16* __restrict__ X, const h16* __restrict__ W,
                            const float* __restrict__ bias, h16* __restrict__ Y)
{
    const int lane  = threadIdx.x & 31;
    const int wave  = blockIdx.x * (blockDim.x >> 5) + (threadIdx.x >> 5);
    const int NB    = G_ / 64;          // 8 column-blocks of 64
    const int mtile = wave / NB;
    const int nblk  = wave % NB;
    const int mrow  = mtile * 16 + (lane & 15);
    const int khalf = (lane < 16) ? 0 : 8;
    const int rowhi = (lane >> 4) << 3; // 0 or 8 (C/D row offset)

    V16 a[4];
#pragma unroll
    for (int kk = 0; kk < 4; ++kk) {
        const h16* p = X + (long)mrow * KP + kk * 32 + khalf;
        a[kk].h[0] = *(const v8h*)(p);
        a[kk].h[1] = *(const v8h*)(p + 16);
    }
#pragma unroll
    for (int nt = 0; nt < 4; ++nt) {
        const int n = nblk * 64 + nt * 16 + (lane & 15);
        v8f c = {};
#pragma unroll
        for (int kk = 0; kk < 4; ++kk) {
            V16 b;
            const h16* p = W + (long)n * KP + kk * 32 + khalf;
            b.h[0] = *(const v8h*)(p);
            b.h[1] = *(const v8h*)(p + 16);
            c = __builtin_amdgcn_wmma_f32_16x16x32_f16(false, a[kk].v, false, b.v,
                                                       (short)0, c, false, false);
        }
        const float bn = bias[n];
#pragma unroll
        for (int r = 0; r < 8; ++r) {
            const int m = mtile * 16 + r + rowhi;
            Y[(long)m * G_ + n] = (h16)(c[r] + bn);
        }
    }
}

// ---------------------------------------------------------------------------
// Recurrent LSTM layer: 16 persistent workgroups, one 16-row batch tile each.
// W_hh (128KB f16), h (f16), c (f32), gate staging (f32), and a TDM-fed
// double-buffered xg tile (2x16KB) all live in LDS (204KB total < 320KB WGP).
// The Tensor Data Mover prefetches step t+1's pre-activation tile while the
// WMMAs for step t run; wave 0 owns TENSORcnt, barriers publish the buffer.
// ---------------------------------------------------------------------------
__global__ void lstm_kernel(const h16* __restrict__ xg,   // [T,B,G] pre-activation
                            const h16* __restrict__ Whh,  // [G,H]
                            h16* __restrict__ hout)       // [T,B,H]
{
    extern __shared__ char smem[];
    h16*   sW = (h16*)smem;                                   // [G][H]  128KB
    h16*   sh = (h16*)(smem + 131072);                        // [16][H]   4KB
    float* sc = (float*)(smem + 131072 + 4096);               // [16][H]   8KB
    float* sg = (float*)(smem + 131072 + 4096 + 8192);        // [16][G]  32KB
    h16*   sx = (h16*)(smem + 176128);                        // [2][16][G] 32KB (TDM dest)

    const int tid  = threadIdx.x;
    const int lane = tid & 31;
    const int w    = tid >> 5;            // wave 0..7 -> gate columns [w*64, w*64+64)
    const int bt   = blockIdx.x;          // batch tile

    const unsigned sx_base = (unsigned)(uintptr_t)(void*)sx;  // LDS byte offset
    const unsigned long long xg_ga =
        (unsigned long long)(uintptr_t)xg + (unsigned long long)bt * 16ull * G_ * 2ull;
    const unsigned long long step_bytes = (unsigned long long)B_ * G_ * 2ull;

    // Prefetch xg tile for t=0 into buffer 0 (wave 0 only; EXEC-independent DMA).
    if (tid < 32) tdm_load_16kb(xg_ga, sx_base);

    for (int e = tid; e < (G_ * H_) / 8; e += 256)
        ((v8h*)sW)[e] = ((const v8h*)Whh)[e];
    for (int e = tid; e < 16 * H_; e += 256) { sh[e] = (h16)0.0f; sc[e] = 0.0f; }

    const int khalf = (lane < 16) ? 0 : 8;
    const int mA    = lane & 15;
    const int rowhi = (lane >> 4) << 3;

    for (int t = 0; t < T_; ++t) {
        if (tid < 32) __builtin_amdgcn_s_wait_tensorcnt(0);   // tile t landed in LDS
        __syncthreads();
        if (tid < 32 && (t + 1) < T_)                          // prefetch tile t+1
            tdm_load_16kb(xg_ga + (unsigned long long)(t + 1) * step_bytes,
                          sx_base + (unsigned)(((t + 1) & 1) * 16384));

        const h16* xs = sx + (t & 1) * (16 * G_);

        V16 a[4];
#pragma unroll
        for (int kk = 0; kk < 4; ++kk) {
            const h16* p = sh + mA * H_ + kk * 32 + khalf;
            a[kk].h[0] = *(const v8h*)(p);
            a[kk].h[1] = *(const v8h*)(p + 16);
        }
#pragma unroll
        for (int nt = 0; nt < 4; ++nt) {
            const int n = w * 64 + nt * 16 + (lane & 15);
            v8f c;
#pragma unroll
            for (int r = 0; r < 8; ++r)
                c[r] = (float)xs[(r + rowhi) * G_ + n];
#pragma unroll
            for (int kk = 0; kk < 4; ++kk) {
                V16 b;
                const h16* p = sW + n * H_ + kk * 32 + khalf;
                b.h[0] = *(const v8h*)(p);
                b.h[1] = *(const v8h*)(p + 16);
                c = __builtin_amdgcn_wmma_f32_16x16x32_f16(false, a[kk].v, false, b.v,
                                                           (short)0, c, false, false);
            }
#pragma unroll
            for (int r = 0; r < 8; ++r)
                sg[(r + rowhi) * G_ + n] = c[r];
        }
        __syncthreads();

#pragma unroll
        for (int i = 0; i < 8; ++i) {          // 16*128 = 2048 elems / 256 threads
            const int e  = tid + i * 256;
            const int b0 = e >> 7, hh = e & (H_ - 1);
            const float gi = sg[b0 * G_ + hh];
            const float gf = sg[b0 * G_ + 128 + hh];
            const float gg = sg[b0 * G_ + 256 + hh];
            const float go = sg[b0 * G_ + 384 + hh];
            const float ig = sigmoidf_(gi);
            const float fg = sigmoidf_(gf);
            const float cg = tanhf(gg);
            const float og = sigmoidf_(go);
            const float cv = fg * sc[e] + ig * cg;
            sc[e] = cv;
            const float hv = og * tanhf(cv);
            sh[e] = (h16)hv;
            hout[((long)t * B_ + bt * 16 + b0) * H_ + hh] = (h16)hv;
        }
        __syncthreads();
    }
}

// ---------------------------------------------------------------------------
// Masked p-norm pooling: pooled[b,h] = (sum_{t<len} |h|^1.5)^(2/3) * len^(-2/3)
// ---------------------------------------------------------------------------
__global__ void pool_kernel(const h16* __restrict__ h1, const int* __restrict__ lens,
                            float* __restrict__ pooled)
{
    const int e = blockIdx.x * blockDim.x + threadIdx.x;
    if (e >= B_ * H_) return;
    const int b = e >> 7, hh = e & (H_ - 1);
    const int len = lens[b];
    float s = 0.0f;
    for (int t = 0; t < len; ++t) {
        const float v = fabsf((float)h1[((long)t * B_ + b) * H_ + hh]);
        s += v * sqrtf(v);                       // |v|^1.5
    }
    pooled[e] = powf(s, 2.0f / 3.0f) * powf((float)len, -2.0f / 3.0f);
}

__global__ void fc1_kernel(const float* __restrict__ pooled, const float* __restrict__ W1,
                           const float* __restrict__ b1, float* __restrict__ act)
{
    const int e = blockIdx.x * blockDim.x + threadIdx.x;
    if (e >= B_ * D1_) return;
    const int b = e / D1_, j = e % D1_;
    float s = b1[j];
    const float* pr = pooled + b * H_;
    const float* wr = W1 + j * H_;
#pragma unroll 4
    for (int k = 0; k < H_; ++k) s += pr[k] * wr[k];
    act[e] = fmaxf(s, 0.0f);
}

__global__ void head_kernel(const float* __restrict__ act, const float* __restrict__ W2,
                            const float* __restrict__ b2, float* __restrict__ out)
{
    const int b = blockIdx.x * blockDim.x + threadIdx.x;
    if (b >= B_) return;
    float logit[OUT_];
    for (int k = 0; k < OUT_; ++k) {
        float s = b2[k];
        const float* ar = act + b * D1_;
        const float* wr = W2 + k * D1_;
        for (int j = 0; j < D1_; ++j) s += ar[j] * wr[j];
        logit[k] = s;
    }
    float m = logit[0];
    for (int k = 1; k < OUT_; ++k) m = fmaxf(m, logit[k]);
    float sum = 0.0f;
    for (int k = 0; k < OUT_; ++k) { logit[k] = __expf(logit[k] - m); sum += logit[k]; }
    const float inv = 1.0f / sum;
    for (int k = 0; k < OUT_; ++k) out[b * OUT_ + k] = logit[k] * inv;
}

// ---------------------------------------------------------------------------
extern "C" void kernel_launch(void* const* d_in, const int* in_sizes, int n_in,
                              void* d_out, int out_size, void* d_ws, size_t ws_size,
                              hipStream_t stream)
{
    (void)in_sizes; (void)n_in; (void)out_size; (void)ws_size;
    const float* x    = (const float*)d_in[0];
    const int*   lens = (const int*)d_in[1];
    const float* Wih0 = (const float*)d_in[2];
    const float* Whh0 = (const float*)d_in[3];
    const float* bih0 = (const float*)d_in[4];
    const float* bhh0 = (const float*)d_in[5];
    const float* Wih1 = (const float*)d_in[6];
    const float* Whh1 = (const float*)d_in[7];
    const float* bih1 = (const float*)d_in[8];
    const float* bhh1 = (const float*)d_in[9];
    const float* W1   = (const float*)d_in[10];
    const float* b1   = (const float*)d_in[11];
    const float* W2   = (const float*)d_in[12];
    const float* b2   = (const float*)d_in[13];
    float* out = (float*)d_out;

    char* ws = (char*)d_ws;
    size_t off = 0;
    auto alloc = [&](size_t bytes) -> char* {
        char* p = ws + off;
        off += (bytes + 255) & ~(size_t)255;
        return p;
    };
    h16*   x16    = (h16*)alloc((size_t)TB * KP * 2);   // 33.5 MB
    h16*   xg     = (h16*)alloc((size_t)TB * G_ * 2);   // 128 MB (reused for both layers)
    h16*   h0     = (h16*)alloc((size_t)TB * H_ * 2);   // 33.5 MB
    h16*   h1     = (h16*)alloc((size_t)TB * H_ * 2);   // 33.5 MB
    h16*   wih0p  = (h16*)alloc((size_t)G_ * KP * 2);
    h16*   whh0p  = (h16*)alloc((size_t)G_ * KP * 2);
    h16*   wih1p  = (h16*)alloc((size_t)G_ * KP * 2);
    h16*   whh1p  = (h16*)alloc((size_t)G_ * KP * 2);
    float* bias0  = (float*)alloc((size_t)G_ * 4);
    float* bias1  = (float*)alloc((size_t)G_ * 4);
    float* pooled = (float*)alloc((size_t)B_ * H_ * 4);
    float* actb   = (float*)alloc((size_t)B_ * D1_ * 4);

    pack_kernel<<<4096, 256, 0, stream>>>(x, Wih0, Whh0, Wih1, Whh1,
                                          bih0, bhh0, bih1, bhh1,
                                          x16, wih0p, whh0p, wih1p, whh1p, bias0, bias1);

    const int proj_blocks = (TB / 16) * (G_ / 64) / 8;            // 8192 blocks of 8 waves
    const size_t lstm_lds = 131072 + 4096 + 8192 + 32768 + 32768; // 204 KB (< 320 KB WGP LDS)

    // Layer 0
    proj_kernel<<<proj_blocks, 256, 0, stream>>>(x16, wih0p, bias0, xg);
    lstm_kernel<<<B_ / 16, 256, lstm_lds, stream>>>(xg, whh0p, h0);
    // Layer 1
    proj_kernel<<<proj_blocks, 256, 0, stream>>>(h0, wih1p, bias1, xg);
    lstm_kernel<<<B_ / 16, 256, lstm_lds, stream>>>(xg, whh1p, h1);
    // Pooling + head
    pool_kernel<<<(B_ * H_) / 256, 256, 0, stream>>>(h1, lens, pooled);
    fc1_kernel<<<(B_ * D1_) / 256, 256, 0, stream>>>(pooled, W1, b1, actb);
    head_kernel<<<1, 256, 0, stream>>>(actb, W2, b2, out);
}
